// NODE_17927193494211
// MI455X (gfx1250) — matrix operand order repaired and checked by reading, not compiled, not run on hardware
//
#include <hip/hip_runtime.h>
#include <hip/hip_bf16.h>

// NODE-style soft decision forest, fused for gfx1250 (MI455X).
// Kernel 1: pack W (1024x256 f32) -> bf16 hi/lo splits in exact WMMA B-operand
//           lane order (one-time, lives in d_ws, stays hot in L2).
// Kernel 2: GEMM (8192x1024x256) via split-bf16 V_WMMA_F32_16X16X32_BF16
//           (hi*hi + hi*lo + lo*hi ~ f32 accuracy), fused bias+sigmoid,
//           tree leaf expansion, leaf-response dot, mean over trees.

typedef __attribute__((ext_vector_type(16))) __bf16        v16bf;
typedef __attribute__((ext_vector_type(8)))  float         v8f;
typedef __attribute__((ext_vector_type(8)))  unsigned int  v8u;

#define BATCH    8192
#define IN_DIM   256
#define N_TREES  128
#define TDEPTH   8
#define NCOLS    (N_TREES * TDEPTH)       // 1024 GEMM columns
#define M_TILE   32                       // batch rows per workgroup (2 sub-tiles)
#define K_STEPS  (IN_DIM / 32)            // 8 WMMA K-steps
#define WAVES    8                        // 256 threads
#define N_TILES  (NCOLS / 16)             // 64
#define TILES_PER_WAVE (N_TILES / WAVES)  // 8
// packed W: [N_TILES][K_STEPS][32 lanes][16 bf16]  (hi then lo)
#define WPACK_ELEMS (N_TILES * K_STEPS * 32 * 16)   // 262144 ushorts = 512 KB
#define WS_NEEDED   (2u * WPACK_ELEMS * sizeof(unsigned short))  // 1 MB

__device__ __forceinline__ unsigned short bf16_rn(float f) {
  unsigned u = __float_as_uint(f);
  u += 0x7FFFu + ((u >> 16) & 1u);      // round-to-nearest-even
  return (unsigned short)(u >> 16);
}
__device__ __forceinline__ float bf16_f32(unsigned short h) {
  return __uint_as_float(((unsigned)h) << 16);
}

// ---------------------------------------------------------------------------
// W pre-pack: one thread = one (ntile, kstep, lane) B-operand fragment (16 bf16)
// lane 0..15 -> column base+lane, K 0..15 of step; lane 16..31 -> K 16..31.
// ---------------------------------------------------------------------------
__global__ __launch_bounds__(256) void pack_w_kernel(
    const float* __restrict__ W, unsigned short* __restrict__ whi,
    unsigned short* __restrict__ wlo) {
  const int id   = blockIdx.x * 256 + threadIdx.x;   // 0 .. 16383
  const int nt   = id >> 8;
  const int s    = (id >> 5) & 7;
  const int lane = id & 31;
  const int n    = nt * 16 + (lane & 15);
  const int kst  = s * 32 + ((lane & 16) ? 16 : 0);
  const float4* wp = (const float4*)(W + (size_t)n * IN_DIM + kst);
  unsigned short h[16], l[16];
#pragma unroll
  for (int q = 0; q < 4; ++q) {
    float4 wv = wp[q];
    float e[4] = {wv.x, wv.y, wv.z, wv.w};
#pragma unroll
    for (int j = 0; j < 4; ++j) {
      h[q * 4 + j] = bf16_rn(e[j]);
      l[q * 4 + j] = bf16_rn(e[j] - bf16_f32(h[q * 4 + j]));
    }
  }
  *(v8u*)(whi + (size_t)id * 16) = *(const v8u*)h;
  *(v8u*)(wlo + (size_t)id * 16) = *(const v8u*)l;
}

// ---------------------------------------------------------------------------
// Main fused kernel. PACKED: B operands streamed pre-split from d_ws.
// ---------------------------------------------------------------------------
template <bool PACKED>
__global__ __launch_bounds__(256) void node_forest_kernel(
    const float* __restrict__ x,     // [8192][256]
    const float* __restrict__ W,     // [1024][256] (fallback path only)
    const float* __restrict__ bias,  // [1024]
    const float* __restrict__ R,     // [128][256]
    const unsigned short* __restrict__ whi,
    const unsigned short* __restrict__ wlo,
    float* __restrict__ out)         // [8192]
{
  // A operand (x tile) staged as bf16 hi/lo in WMMA 16x32 A layout:
  // lane m      : slots 0..15 -> K 0..15 (contiguous K)
  // lane m + 16 : slots 0..15 -> K 16..31
  __shared__ __attribute__((aligned(32))) unsigned short A_lds[2][K_STEPS][2][32][16]; // 32 KB
  __shared__ float probs[WAVES][16][17];  // per-wave 16x16 tile transpose buffer
  __shared__ float red[M_TILE];           // per-row tree-sum reduction

  const int tid  = threadIdx.x;
  const int lane = tid & 31;
  const int wave = tid >> 5;
  const int row_base = blockIdx.x * M_TILE;

  if (tid < M_TILE) red[tid] = 0.0f;

  // ---------- stage x tile (32 rows) -> bf16 hi/lo in LDS ----------
#pragma unroll
  for (int task = tid; task < 512; task += 256) {
    const int m  = task >> 4;        // row 0..31
    const int k0 = (task & 15) * 16; // 16 contiguous K per task
    const int st = m >> 4;           // sub-tile
    const int ml = m & 15;
    const float* xr = x + (size_t)(row_base + m) * IN_DIM + k0;
#pragma unroll
    for (int q = 0; q < 4; ++q) {
      float4 xv = *(const float4*)(xr + q * 4);
      float e[4] = {xv.x, xv.y, xv.z, xv.w};
#pragma unroll
      for (int j = 0; j < 4; ++j) {
        const int k    = k0 + q * 4 + j;
        const int s    = k >> 5;
        const int kk   = k & 31;
        const int dl   = ml + ((kk & 16) ? 16 : 0);
        const int slot = kk & 15;
        const unsigned short h = bf16_rn(e[j]);
        const unsigned short l = bf16_rn(e[j] - bf16_f32(h));
        A_lds[0][s][st][dl][slot] = h;
        A_lds[1][s][st][dl][slot] = l;
      }
    }
  }
  __syncthreads();

  const int col  = lane & 15;  // N within tile / C column
  const int half = lane >> 4;
  const int m_ev = lane & 15;  // tree-eval row within sub-tile
  const int tl   = lane >> 4;  // tree-eval local tree (2 trees per tile)

  float acc_row0 = 0.0f, acc_row1 = 0.0f;

  for (int tt = 0; tt < TILES_PER_WAVE; ++tt) {
    const int nt    = wave + tt * WAVES;  // N-tile 0..63
    const int nbase = nt * 16;

    v8f c0 = {}, c1 = {};
    const float* wrow = W + (size_t)(nbase + col) * IN_DIM + half * 16;
    const unsigned short* ph = whi + ((size_t)(nt * K_STEPS) * 32 + lane) * 16;
    const unsigned short* pl = wlo + ((size_t)(nt * K_STEPS) * 32 + lane) * 16;

    if (PACKED && tt + 1 < TILES_PER_WAVE) {   // prefetch next tile's packed W
      __builtin_prefetch(whi + ((size_t)((nt + WAVES) * K_STEPS) * 32 + lane) * 16, 0, 1);
      __builtin_prefetch(wlo + ((size_t)((nt + WAVES) * K_STEPS) * 32 + lane) * 16, 0, 1);
    }

#pragma unroll
    for (int s = 0; s < K_STEPS; ++s) {
      v16bf Bh, Bl;
      if (PACKED) {
        Bh = *(const v16bf*)(ph + (size_t)s * 32 * 16);
        Bl = *(const v16bf*)(pl + (size_t)s * 32 * 16);
      } else {
        const float4* wp = (const float4*)(wrow + s * 32);
        v8u bhu, blu;
#pragma unroll
        for (int q = 0; q < 4; ++q) {
          float4 wv = wp[q];
          float e[4] = {wv.x, wv.y, wv.z, wv.w};
          unsigned short hh[4], ll[4];
#pragma unroll
          for (int j = 0; j < 4; ++j) {
            hh[j] = bf16_rn(e[j]);
            ll[j] = bf16_rn(e[j] - bf16_f32(hh[j]));
          }
          bhu[q * 2 + 0] = (unsigned)hh[0] | ((unsigned)hh[1] << 16);
          bhu[q * 2 + 1] = (unsigned)hh[2] | ((unsigned)hh[3] << 16);
          blu[q * 2 + 0] = (unsigned)ll[0] | ((unsigned)ll[1] << 16);
          blu[q * 2 + 1] = (unsigned)ll[2] | ((unsigned)ll[3] << 16);
        }
        Bh = __builtin_bit_cast(v16bf, bhu);
        Bl = __builtin_bit_cast(v16bf, blu);
      }
      v16bf Ah0 = *(const v16bf*)(&A_lds[0][s][0][lane][0]);
      v16bf Al0 = *(const v16bf*)(&A_lds[1][s][0][lane][0]);
      v16bf Ah1 = *(const v16bf*)(&A_lds[0][s][1][lane][0]);
      v16bf Al1 = *(const v16bf*)(&A_lds[1][s][1][lane][0]);
      // D = A*B + C ; split-bf16: hi*hi + hi*lo + lo*hi ; B shared by 2 M-tiles
      c0 = __builtin_amdgcn_wmma_f32_16x16x32_bf16(false, Ah0, false, Bh, (short)0, c0, false, false);
      c1 = __builtin_amdgcn_wmma_f32_16x16x32_bf16(false, Ah1, false, Bh, (short)0, c1, false, false);
      c0 = __builtin_amdgcn_wmma_f32_16x16x32_bf16(false, Ah0, false, Bl, (short)0, c0, false, false);
      c1 = __builtin_amdgcn_wmma_f32_16x16x32_bf16(false, Ah1, false, Bl, (short)0, c1, false, false);
      c0 = __builtin_amdgcn_wmma_f32_16x16x32_bf16(false, Al0, false, Bh, (short)0, c0, false, false);
      c1 = __builtin_amdgcn_wmma_f32_16x16x32_bf16(false, Al1, false, Bh, (short)0, c1, false, false);
    }

    const float bcol = bias[nbase + col];
    const int   tg   = nt * 2 + tl;
    const float4* R4 = (const float4*)(R + (size_t)tg * 256);

#pragma unroll
    for (int sub = 0; sub < 2; ++sub) {
      const v8f& c = sub ? c1 : c0;
      // ---- bias + sigmoid, transpose sub-tile through per-wave LDS ----
#pragma unroll
      for (int i = 0; i < 8; ++i) {
        const float z = c[i] + bcol;               // C layout: lane col, M = half*8+i
        const float p = 1.0f / (1.0f + __expf(-z));
        probs[wave][half * 8 + i][col] = p;
      }
      asm volatile("s_wait_dscnt 0" ::: "memory"); // wave-internal LDS RAW

      // ---- tree evaluation: lane -> (row m_ev, local tree tl) ----
      float p[TDEPTH];
#pragma unroll
      for (int d = 0; d < TDEPTH; ++d) p[d] = probs[wave][m_ev][tl * 8 + d];

      // leaf l = (depth0..3 as MSBs)*16 + (depth4..7); weight = u[i]*v[j]
      float u[16], v[16];
      u[0] = 1.0f; v[0] = 1.0f;
#pragma unroll
      for (int d = 0; d < 4; ++d) {
        const int cnt = 1 << d;
#pragma unroll
        for (int i = cnt - 1; i >= 0; --i) {
          const float tu = u[i], tv = v[i];
          u[2 * i + 1] = tu * p[d];     u[2 * i] = tu - u[2 * i + 1];
          v[2 * i + 1] = tv * p[4 + d]; v[2 * i] = tv - v[2 * i + 1];
        }
      }
      float tsum = 0.0f;
#pragma unroll 4
      for (int i = 0; i < 16; ++i) {
        float ssum = 0.0f;
#pragma unroll
        for (int q = 0; q < 4; ++q) {
          const float4 r = R4[i * 4 + q];
          ssum = fmaf(v[q * 4 + 0], r.x, ssum);
          ssum = fmaf(v[q * 4 + 1], r.y, ssum);
          ssum = fmaf(v[q * 4 + 2], r.z, ssum);
          ssum = fmaf(v[q * 4 + 3], r.w, ssum);
        }
        tsum = fmaf(u[i], ssum, tsum);
      }
      if (sub) acc_row1 += tsum; else acc_row0 += tsum;
      asm volatile("s_wait_dscnt 0" ::: "memory"); // reads done before buffer reuse
    }
  }

  // ---------- reduce over trees (lanes + waves), mean, store ----------
  atomicAdd(&red[m_ev], acc_row0);        // ds_add_f32, rows 0..15
  atomicAdd(&red[16 + m_ev], acc_row1);   // rows 16..31
  __syncthreads();
  if (tid < M_TILE) out[row_base + tid] = red[tid] * (1.0f / (float)N_TREES);
}

extern "C" void kernel_launch(void* const* d_in, const int* in_sizes, int n_in,
                              void* d_out, int out_size, void* d_ws, size_t ws_size,
                              hipStream_t stream) {
  (void)in_sizes; (void)n_in; (void)out_size;
  const float* x    = (const float*)d_in[0];  // (8192, 256)
  const float* W    = (const float*)d_in[1];  // (128, 8, 256)
  const float* bias = (const float*)d_in[2];  // (128, 8)
  const float* R    = (const float*)d_in[3];  // (128, 256, 1)
  float* out        = (float*)d_out;          // (8192,)

  dim3 block(256);
  dim3 grid(BATCH / M_TILE);                  // 256 workgroups

  if (d_ws != nullptr && ws_size >= WS_NEEDED) {
    unsigned short* whi = (unsigned short*)d_ws;
    unsigned short* wlo = whi + WPACK_ELEMS;
    pack_w_kernel<<<dim3((N_TILES * K_STEPS * 32) / 256), block, 0, stream>>>(W, whi, wlo);
    node_forest_kernel<true><<<grid, block, 0, stream>>>(x, W, bias, R, whi, wlo, out);
  } else {
    node_forest_kernel<false><<<grid, block, 0, stream>>>(x, W, bias, R, nullptr, nullptr, out);
  }
}
